// CSLSTM_87136296501667
// MI455X (gfx1250) — 8-bit matrix operand order repaired and checked by reading, compiled, not run on hardware
//
#include <hip/hip_runtime.h>

#define B_TOT   16384
#define HDIM    128
#define T_INN   30
#define T_OUTN  50
#define ROWS    64
#define MT      4

typedef int   v16i __attribute__((ext_vector_type(16)));
typedef float v8f  __attribute__((ext_vector_type(8)));

#ifndef __has_builtin
#define __has_builtin(x) 0
#endif

__device__ __forceinline__ unsigned f2fp8(float x) {
#if __has_builtin(__builtin_amdgcn_cvt_pk_fp8_f32)
  return (unsigned)__builtin_amdgcn_cvt_pk_fp8_f32(x, 0.f, 0, false) & 0xffu;
#else
  unsigned u = __float_as_uint(x);
  unsigned s = (u >> 24) & 0x80u;
  unsigned a = u & 0x7fffffffu;
  a = (a > 0x43E00000u) ? 0x43E00000u : a;
  if (a < 0x3C800000u) return s;
  a += 0x00080000u;
  a = (a > 0x43E00000u) ? 0x43E00000u : a;
  return s | ((a >> 20) - 960u);
#endif
}

__device__ __forceinline__ int pack_fp8x4(float a, float b, float c, float d) {
#if __has_builtin(__builtin_amdgcn_cvt_pk_fp8_f32)
  int r = __builtin_amdgcn_cvt_pk_fp8_f32(a, b, 0, false);
  r = __builtin_amdgcn_cvt_pk_fp8_f32(c, d, r, true);
  return r;
#else
  return (int)(f2fp8(a) | (f2fp8(b) << 8) | (f2fp8(c) << 16) | (f2fp8(d) << 24));
#endif
}

__device__ __forceinline__ float tanh_hw(float x) {
#if __has_builtin(__builtin_amdgcn_tanhf)
  return __builtin_amdgcn_tanhf(x);
#else
  float e = __expf(x + x);
  return 1.0f - 2.0f * __builtin_amdgcn_rcpf(e + 1.0f);
#endif
}
__device__ __forceinline__ float sigf(float x) {
  return fmaf(0.5f, tanh_hw(0.5f * x), 0.5f);
}

__device__ __forceinline__ v16i load_bfrag(const float* __restrict__ W, int n0,
                                           int l16, int lhalf) {
  const float* wr = W + (size_t)(n0 + l16) * HDIM;
  v16i b;
#pragma unroll
  for (int v = 0; v < 16; ++v) {
    int kb = (v >> 3) * 64 + ((v >> 1) & 3) * 16 + lhalf * 8 + (v & 1) * 4;
    b[v] = pack_fp8x4(wr[kb + 0], wr[kb + 1], wr[kb + 2], wr[kb + 3]);
  }
  return b;
}

__global__ __launch_bounds__(256)
void lstm_s2s_fused(const float* __restrict__ x,
                    const float* __restrict__ eWih0, const float* __restrict__ eWhh0,
                    const float* __restrict__ eb0,
                    const float* __restrict__ eWih1, const float* __restrict__ eWhh1,
                    const float* __restrict__ eb1,
                    const float* __restrict__ dWih0, const float* __restrict__ dWhh0,
                    const float* __restrict__ db0,
                    const float* __restrict__ dWih1, const float* __restrict__ dWhh1,
                    const float* __restrict__ db1,
                    const float* __restrict__ fcW, const float* __restrict__ fcb,
                    float* __restrict__ out)
{
  __shared__ __align__(16) unsigned int hbuf[4][MT][32][16];
  __shared__ float pr[2][ROWS][2];
  __shared__ float sx0[ROWS], sx1[ROWS];

  const int tid   = threadIdx.x;
  const int wave  = tid >> 5;
  const int lane  = tid & 31;
  const int l16   = lane & 15;
  const int lhalf = lane >> 4;
  const int col   = wave * 16 + l16;
  const int rowB  = blockIdx.x * ROWS;

  const int hsel    = (col >> 3) & 1;
  const int c64     = col & 63;
  const int dw      = ((col >> 6) << 3) + ((c64 >> 4) << 1) + ((c64 & 7) >> 2);
  const int byteIdx = col & 3;
  unsigned char* hb8 = (unsigned char*)hbuf;

  for (int i = tid; i < 4 * MT * 32 * 16; i += 256) ((unsigned int*)hbuf)[i] = 0u;

  v16i bHH0[4], bIH1[4], bHH1[4];
  float bb0[4], w0[4], w1[4], bb1[4];
#pragma unroll
  for (int g = 0; g < 4; ++g) {
    int n0 = g * HDIM + wave * 16;
    bHH0[g] = load_bfrag(eWhh0, n0, l16, lhalf);
    bIH1[g] = load_bfrag(eWih1, n0, l16, lhalf);
    bHH1[g] = load_bfrag(eWhh1, n0, l16, lhalf);
    bb0[g] = eb0[n0 + l16];
    w0[g]  = eWih0[(n0 + l16) * 2 + 0];
    w1[g]  = eWih0[(n0 + l16) * 2 + 1];
    bb1[g] = eb1[n0 + l16];
  }

  float cReg[2][MT][8];
#pragma unroll
  for (int L = 0; L < 2; ++L)
#pragma unroll
    for (int m = 0; m < MT; ++m)
#pragma unroll
      for (int r = 0; r < 8; ++r) cReg[L][m][r] = 0.f;

  __syncthreads();
  int ping = 0;

#pragma unroll 1
  for (int t = 0; t < T_INN; ++t) {
    if (tid < ROWS) {
      int row = rowB + tid;
      sx0[tid] = x[row * 60 + t];
      sx1[tid] = x[row * 60 + 30 + t];
    }
    __syncthreads();
#pragma unroll
    for (int mt = 0; mt < MT; ++mt) {
      const unsigned int* ap = &hbuf[ping][mt][lane][0];
      v16i aF;
#pragma unroll
      for (int i = 0; i < 16; ++i) aF[i] = (int)ap[i];
      v8f acc[4];
#pragma unroll
      for (int g = 0; g < 4; ++g) {
        v8f ci;
#pragma unroll
        for (int r = 0; r < 8; ++r) {
          int m = mt * 16 + r + 8 * lhalf;
          ci[r] = bb0[g] + sx0[m] * w0[g] + sx1[m] * w1[g];
        }
        acc[g] = __builtin_amdgcn_wmma_f32_16x16x128_fp8_fp8(aF, bHH0[g], (short)0, ci, false, false);
      }
#pragma unroll
      for (int r = 0; r < 8; ++r) {
        float iv = sigf(acc[0][r]), fv = sigf(acc[1][r]);
        float gv = tanh_hw(acc[2][r]), ov = sigf(acc[3][r]);
        float c = fv * cReg[0][mt][r] + iv * gv;
        cReg[0][mt][r] = c;
        float h = ov * tanh_hw(c);
        int lane2 = r + 8 * lhalf + 16 * hsel;
        hb8[(((((1 - ping) * MT + mt) * 32 + lane2) * 16 + dw) << 2) + byteIdx] = (unsigned char)f2fp8(h);
      }
    }
    __syncthreads();
#pragma unroll
    for (int mt = 0; mt < MT; ++mt) {
      const unsigned int* a0p = &hbuf[1 - ping][mt][lane][0];
      const unsigned int* a1p = &hbuf[2 + ping][mt][lane][0];
      v16i a0, a1;
#pragma unroll
      for (int i = 0; i < 16; ++i) { a0[i] = (int)a0p[i]; a1[i] = (int)a1p[i]; }
      v8f acc[4];
#pragma unroll
      for (int g = 0; g < 4; ++g) {
        v8f ci;
#pragma unroll
        for (int r = 0; r < 8; ++r) ci[r] = bb1[g];
        v8f tv = __builtin_amdgcn_wmma_f32_16x16x128_fp8_fp8(a0, bIH1[g], (short)0, ci, false, false);
        acc[g] = __builtin_amdgcn_wmma_f32_16x16x128_fp8_fp8(a1, bHH1[g], (short)0, tv, false, false);
      }
#pragma unroll
      for (int r = 0; r < 8; ++r) {
        float iv = sigf(acc[0][r]), fv = sigf(acc[1][r]);
        float gv = tanh_hw(acc[2][r]), ov = sigf(acc[3][r]);
        float c = fv * cReg[1][mt][r] + iv * gv;
        cReg[1][mt][r] = c;
        float h = ov * tanh_hw(c);
        int lane2 = r + 8 * lhalf + 16 * hsel;
        hb8[((((3 - ping) * MT + mt) * 32 + lane2) * 16 + dw) * 4 + byteIdx] = (unsigned char)f2fp8(h);
      }
    }
    __syncthreads();
    ping ^= 1;
  }

#pragma unroll
  for (int g = 0; g < 4; ++g) {
    int n0 = g * HDIM + wave * 16;
    bHH0[g] = load_bfrag(dWhh0, n0, l16, lhalf);
    bIH1[g] = load_bfrag(dWih1, n0, l16, lhalf);
    bHH1[g] = load_bfrag(dWhh1, n0, l16, lhalf);
    bb0[g] = db0[n0 + l16];
    w0[g]  = dWih0[(n0 + l16) * 2 + 0];
    w1[g]  = dWih0[(n0 + l16) * 2 + 1];
    bb1[g] = db1[n0 + l16];
  }
  const float fw0 = fcW[col], fw1 = fcW[HDIM + col];
  const float fcbv = fcb[tid & 1];

  if (tid < ROWS) {
    int row = rowB + tid;
    pr[0][tid][0] = x[row * 60 + (T_INN - 1)];
    pr[0][tid][1] = x[row * 60 + 30 + (T_INN - 1)];
  }
  __syncthreads();
  int pp = 0;

#pragma unroll 1
  for (int t = 0; t < T_OUTN; ++t) {
    if (tid < 2 * ROWS) ((float*)pr[1 - pp])[tid] = 0.f;
#pragma unroll
    for (int mt = 0; mt < MT; ++mt) {
      const unsigned int* ap = &hbuf[ping][mt][lane][0];
      v16i aF;
#pragma unroll
      for (int i = 0; i < 16; ++i) aF[i] = (int)ap[i];
      v8f acc[4];
#pragma unroll
      for (int g = 0; g < 4; ++g) {
        v8f ci;
#pragma unroll
        for (int r = 0; r < 8; ++r) {
          int m = mt * 16 + r + 8 * lhalf;
          ci[r] = bb0[g] + pr[pp][m][0] * w0[g] + pr[pp][m][1] * w1[g];
        }
        acc[g] = __builtin_amdgcn_wmma_f32_16x16x128_fp8_fp8(aF, bHH0[g], (short)0, ci, false, false);
      }
#pragma unroll
      for (int r = 0; r < 8; ++r) {
        float iv = sigf(acc[0][r]), fv = sigf(acc[1][r]);
        float gv = tanh_hw(acc[2][r]), ov = sigf(acc[3][r]);
        float c = fv * cReg[0][mt][r] + iv * gv;
        cReg[0][mt][r] = c;
        float h = ov * tanh_hw(c);
        int lane2 = r + 8 * lhalf + 16 * hsel;
        hb8[(((((1 - ping) * MT + mt) * 32 + lane2) * 16 + dw) << 2) + byteIdx] = (unsigned char)f2fp8(h);
      }
    }
    __syncthreads();
#pragma unroll
    for (int mt = 0; mt < MT; ++mt) {
      const unsigned int* a0p = &hbuf[1 - ping][mt][lane][0];
      const unsigned int* a1p = &hbuf[2 + ping][mt][lane][0];
      v16i a0, a1;
#pragma unroll
      for (int i = 0; i < 16; ++i) { a0[i] = (int)a0p[i]; a1[i] = (int)a1p[i]; }
      v8f acc[4];
#pragma unroll
      for (int g = 0; g < 4; ++g) {
        v8f ci;
#pragma unroll
        for (int r = 0; r < 8; ++r) ci[r] = bb1[g];
        v8f tv = __builtin_amdgcn_wmma_f32_16x16x128_fp8_fp8(a0, bIH1[g], (short)0, ci, false, false);
        acc[g] = __builtin_amdgcn_wmma_f32_16x16x128_fp8_fp8(a1, bHH1[g], (short)0, tv, false, false);
      }
#pragma unroll
      for (int r = 0; r < 8; ++r) {
        float iv = sigf(acc[0][r]), fv = sigf(acc[1][r]);
        float gv = tanh_hw(acc[2][r]), ov = sigf(acc[3][r]);
        float c = fv * cReg[1][mt][r] + iv * gv;
        cReg[1][mt][r] = c;
        float h = ov * tanh_hw(c);
        int lane2 = r + 8 * lhalf + 16 * hsel;
        hb8[((((3 - ping) * MT + mt) * 32 + lane2) * 16 + dw) * 4 + byteIdx] = (unsigned char)f2fp8(h);
        float t0 = h * fw0, t1 = h * fw1;
#pragma unroll
        for (int msk = 1; msk < 16; msk <<= 1) {
          t0 += __shfl_xor(t0, msk, 32);
          t1 += __shfl_xor(t1, msk, 32);
        }
        if (l16 == 0) {
          int m = mt * 16 + r + 8 * lhalf;
          atomicAdd(&pr[1 - pp][m][0], t0);
          atomicAdd(&pr[1 - pp][m][1], t1);
        }
      }
    }
    __syncthreads();
    if (tid < 2 * ROWS) {
      int m = tid >> 1, f = tid & 1;
      float v = pr[1 - pp][m][f] + fcbv;
      pr[1 - pp][m][f] = v;
      out[(size_t)(rowB + m) * (T_OUTN * 2) + t * 2 + f] = v;
    }
    __syncthreads();
    ping ^= 1; pp ^= 1;
  }
}

extern "C" void kernel_launch(void* const* d_in, const int* in_sizes, int n_in,
                              void* d_out, int out_size, void* d_ws, size_t ws_size,
                              hipStream_t stream) {
  (void)in_sizes; (void)n_in; (void)out_size; (void)d_ws; (void)ws_size;
  const float* x     = (const float*)d_in[0];
  const float* eWih0 = (const float*)d_in[1];
  const float* eWhh0 = (const float*)d_in[2];
  const float* eb0   = (const float*)d_in[3];
  const float* eWih1 = (const float*)d_in[4];
  const float* eWhh1 = (const float*)d_in[5];
  const float* eb1   = (const float*)d_in[6];
  const float* dWih0 = (const float*)d_in[7];
  const float* dWhh0 = (const float*)d_in[8];
  const float* db0   = (const float*)d_in[9];
  const float* dWih1 = (const float*)d_in[10];
  const float* dWhh1 = (const float*)d_in[11];
  const float* db1   = (const float*)d_in[12];
  const float* fcW   = (const float*)d_in[13];
  const float* fcb   = (const float*)d_in[14];

  dim3 grid(B_TOT / ROWS), block(256);
  hipLaunchKernelGGL(lstm_s2s_fused, grid, block, 0, stream,
                     x, eWih0, eWhh0, eb0, eWih1, eWhh1, eb1,
                     dWih0, dWhh0, db0, dWih1, dWhh1, db1,
                     fcW, fcb, (float*)d_out);
}
